// CustomModel_64647847739444
// MI455X (gfx1250) — compile-verified, matrix operand order checked by reading, not run on hardware
//
#include <hip/hip_runtime.h>
#include <hip/hip_bf16.h>
#include <math.h>

// ---------------------------------------------------------------------------
// 3-layer bidirectional GRU (Keras reset_after=True, gate order z,r,h)
// B=256, T=512, D=256, H=128, 3H=384.  bf16 WMMA for both the input
// projections (big parallel GEMM) and the recurrent h@U per-step GEMM.
// ---------------------------------------------------------------------------

#define HU     128      // GRU units
#define BATCH  256
#define TSEQ   512
#define DIN    256      // input feature dim of every layer (D == 2H == 256)
#define NOUT   384      // 3H
#define UPAD   136      // U / h LDS row pad (bf16): dword stride 68 -> no conflicts
#define RPAD   388      // rec LDS row pad (f32)

typedef __attribute__((ext_vector_type(16))) __bf16 v16bf;
typedef __attribute__((ext_vector_type(8)))  __bf16 v8bf;
typedef __attribute__((ext_vector_type(8)))  float  v8f;

static __device__ __forceinline__ v16bf frag_cat(v8bf lo, v8bf hi) {
  return __builtin_shufflevector(lo, hi, 0,1,2,3,4,5,6,7,8,9,10,11,12,13,14,15);
}
static __device__ __forceinline__ float sigmoidf(float x) {
  return 1.0f / (1.0f + __expf(-x));
}

// --------------------------- weight conversion -----------------------------
// Wt[ld][n][k] = (bf16) Ws[ld][k][n]   (ld = layer*2+dir, k<256, n<384)
// Ut[ld][n][k] = (bf16) Us[ld][k][n]   (k<128, n<384)
__global__ void conv_weights(const float* __restrict__ Ws,
                             const float* __restrict__ Us,
                             __bf16* __restrict__ Wt,
                             __bf16* __restrict__ Ut) {
  const int stride = gridDim.x * blockDim.x;
  const int totW = 6 * DIN * NOUT;
  for (int i = blockIdx.x * blockDim.x + threadIdx.x; i < totW; i += stride) {
    int ld = i / (DIN * NOUT);
    int r  = i % (DIN * NOUT);
    int n  = r / DIN;
    int k  = r % DIN;
    Wt[i] = (__bf16)Ws[(size_t)ld * DIN * NOUT + (size_t)k * NOUT + n];
  }
  const int totU = 6 * HU * NOUT;
  for (int i = blockIdx.x * blockDim.x + threadIdx.x; i < totU; i += stride) {
    int ld = i / (HU * NOUT);
    int r  = i % (HU * NOUT);
    int n  = r / HU;
    int k  = r % HU;
    Ut[i] = (__bf16)Us[(size_t)ld * HU * NOUT + (size_t)k * NOUT + n];
  }
}

__global__ void conv_f32_bf16(const float* __restrict__ in,
                              __bf16* __restrict__ out, size_t n) {
  size_t stride = (size_t)gridDim.x * blockDim.x;
  for (size_t i = blockIdx.x * (size_t)blockDim.x + threadIdx.x; i < n; i += stride)
    out[i] = (__bf16)in[i];
}

// --------------------------- input projection GEMM -------------------------
// xp[m][n] = sum_k seq[m][k] * Wt[n][k] + b_i[n],  m = b*T+t  (bf16 out)
// Per wave: one 16(M) x 64(N) tile, K=256 in 8 chunks of x32,
// software-pipelined (next chunk's fragments load under current WMMAs).
__global__ void xp_gemm(const __bf16* __restrict__ seq,
                        const __bf16* __restrict__ Wt_all,
                        const float* __restrict__ bs, int layer,
                        __bf16* __restrict__ xp_f, __bf16* __restrict__ xp_b) {
  const int dir = blockIdx.y;
  const __bf16* __restrict__ Wt = Wt_all + (size_t)(layer * 2 + dir) * NOUT * DIN;
  const float* __restrict__ bi  = bs + ((size_t)(layer * 2 + dir) * 2 + 0) * NOUT;
  __bf16* __restrict__ xp = dir ? xp_b : xp_f;

  const int warp = threadIdx.x >> 5;
  const int lane = threadIdx.x & 31;
  const int idx  = lane & 15;
  const int hs   = lane >> 4;

  const int gw = blockIdx.x * 8 + warp;   // 49152 waves total
  const int mt = gw / 6;                  // 0..8191  (M tile of 16 rows)
  const int ns = gw % 6;                  // N strip of 4 tiles (64 cols)
  const size_t m0 = (size_t)mt * 16;

  const __bf16* arow_base = seq + (m0 + idx) * DIN + hs * 8;
  const __bf16* brow_base = Wt + (size_t)((ns * 4) * 16 + idx) * DIN + hs * 8;

  v8f acc[4];
#pragma unroll
  for (int j = 0; j < 4; j++) acc[j] = (v8f)(0.0f);

  // prologue: stage 0 fragments
  v16bf a_cur, b_cur[4];
  {
    const v8bf* ar = (const v8bf*)(arow_base);
    a_cur = frag_cat(ar[0], ar[2]);
#pragma unroll
    for (int j = 0; j < 4; j++) {
      const v8bf* br = (const v8bf*)(brow_base + (size_t)j * 16 * DIN);
      b_cur[j] = frag_cat(br[0], br[2]);
    }
  }
#pragma unroll
  for (int kc = 0; kc < 8; kc++) {
    v16bf a_nxt, b_nxt[4];
    if (kc < 7) {                      // issue next chunk's loads first
      const int k1 = (kc + 1) * 32;
      const v8bf* ar = (const v8bf*)(arow_base + k1);
      a_nxt = frag_cat(ar[0], ar[2]);
#pragma unroll
      for (int j = 0; j < 4; j++) {
        const v8bf* br = (const v8bf*)(brow_base + (size_t)j * 16 * DIN + k1);
        b_nxt[j] = frag_cat(br[0], br[2]);
      }
    }
#pragma unroll
    for (int j = 0; j < 4; j++)
      acc[j] = __builtin_amdgcn_wmma_f32_16x16x32_bf16(
          false, a_cur, false, b_cur[j], (short)0, acc[j], false, false);
    if (kc < 7) {
      a_cur = a_nxt;
#pragma unroll
      for (int j = 0; j < 4; j++) b_cur[j] = b_nxt[j];
    }
  }
#pragma unroll
  for (int j = 0; j < 4; j++) {
    const int n = (ns * 4 + j) * 16 + idx;
    const float bias = bi[n];
#pragma unroll
    for (int v = 0; v < 8; v++) {
      size_t row = m0 + v + hs * 8;
      xp[row * NOUT + n] = (__bf16)(acc[j][v] + bias);
    }
  }
}

// --------------------------- recurrent scan --------------------------------
// One block = 16 batch rows, one direction. 8 waves x 3 N-tiles cover N=384.
// U (bf16) staged to LDS via async global->LDS DMA, B-fragments preloaded to
// VGPRs (loop-invariant). h kept bf16 in LDS (WMMA A operand) + f32 in the
// gate threads' registers (full-precision blend). rec exchanged through LDS.
__global__ void gru_scan(const __bf16* __restrict__ xp_f,
                         const __bf16* __restrict__ xp_b,
                         const __bf16* __restrict__ Ut_all,
                         const float* __restrict__ bs, int layer,
                         __bf16* __restrict__ out_seq) {
  extern __shared__ char smem[];
  __bf16* U_lds = (__bf16*)smem;                                    // [384][UPAD]
  __bf16* h_bf  = (__bf16*)(smem + (size_t)NOUT * UPAD * sizeof(__bf16)); // [16][UPAD]
  float*  rec   = (float*)(smem + (size_t)(NOUT + 16) * UPAD * sizeof(__bf16)); // [16][RPAD]

  const int dir = blockIdx.y;
  const int b0  = blockIdx.x * 16;
  const __bf16* __restrict__ xp = dir ? xp_b : xp_f;
  const __bf16* __restrict__ Ut = Ut_all + (size_t)(layer * 2 + dir) * NOUT * HU;
  const float* __restrict__ bh  = bs + ((size_t)(layer * 2 + dir) * 2 + 1) * NOUT;
  const int tid = threadIdx.x;

  // stage U into padded LDS: per-lane 16B async global->LDS DMA (ASYNCcnt)
  for (int i = tid; i < NOUT * 16; i += 256) {
    int r = i >> 4, g = i & 15;
    unsigned ldsoff = (unsigned)(uintptr_t)(U_lds + r * UPAD + g * 8);
    const __bf16* gsrc = Ut + r * HU + g * 8;
    asm volatile("global_load_async_to_lds_b128 %0, %1, off"
                 :: "v"(ldsoff), "v"(gsrc) : "memory");
  }
  // zero the bf16 h tile
  for (int i = tid; i < 16 * UPAD; i += 256) h_bf[i] = (__bf16)0.0f;
  asm volatile("s_wait_asynccnt 0x0" ::: "memory");
  __syncthreads();

  const int warp = tid >> 5, lane = tid & 31;
  const int idx  = lane & 15, hs = lane >> 4;

  // loop-invariant B fragments of U: 3 N-tiles x 4 K-chunks
  v16bf bfrag[3][4];
#pragma unroll
  for (int j = 0; j < 3; j++) {
    const int n = (warp * 3 + j) * 16 + idx;
#pragma unroll
    for (int kc = 0; kc < 4; kc++) {
      const v8bf* p = (const v8bf*)(U_lds + n * UPAD + kc * 32 + hs * 8);
      bfrag[j][kc] = frag_cat(p[0], p[2]);
    }
  }

  // gate-phase mapping: row = tid>>4 (batch row), 8 units per thread
  const int grow = tid >> 4;
  const int ucol = (tid & 15) * 8;
  const int b    = b0 + grow;
  float bhz[8], bhr[8], bhh[8], hreg[8];
#pragma unroll
  for (int e = 0; e < 8; e++) {
    bhz[e] = bh[ucol + e];
    bhr[e] = bh[128 + ucol + e];
    bhh[e] = bh[256 + ucol + e];
    hreg[e] = 0.0f;
  }

  for (int step = 0; step < TSEQ; step++) {
    const int t = dir ? (TSEQ - 1 - step) : step;

    // issue this step's xp loads early: they drain under the WMMAs + barrier
    const size_t xbase = ((size_t)b * TSEQ + t) * NOUT;
    const v8bf xz8 = *(const v8bf*)(xp + xbase + ucol);
    const v8bf xr8 = *(const v8bf*)(xp + xbase + 128 + ucol);
    const v8bf xh8 = *(const v8bf*)(xp + xbase + 256 + ucol);

    // ---- phase 1: rec = h @ U ----
    v8f acc[3];
#pragma unroll
    for (int j = 0; j < 3; j++) acc[j] = (v8f)(0.0f);
#pragma unroll
    for (int kc = 0; kc < 4; kc++) {
      const v8bf* ap = (const v8bf*)(h_bf + idx * UPAD + kc * 32 + hs * 8);
      v16bf a = frag_cat(ap[0], ap[2]);
#pragma unroll
      for (int j = 0; j < 3; j++)
        acc[j] = __builtin_amdgcn_wmma_f32_16x16x32_bf16(
            false, a, false, bfrag[j][kc], (short)0, acc[j], false, false);
    }
#pragma unroll
    for (int j = 0; j < 3; j++) {
      const int col = (warp * 3 + j) * 16 + idx;
#pragma unroll
      for (int v = 0; v < 8; v++)
        rec[(v + hs * 8) * RPAD + col] = acc[j][v];
    }
    __syncthreads();   // rec complete; all A reads of h done

    // ---- phase 2: gates + h update ----
    {
      if (step + 1 < TSEQ) {   // prefetch next timestep's xp
        const int tn = dir ? (TSEQ - 2 - step) : (step + 1);
        const size_t xb2 = ((size_t)b * TSEQ + tn) * NOUT;
        __builtin_prefetch(xp + xb2 + ucol, 0, 0);
        __builtin_prefetch(xp + xb2 + 128 + ucol, 0, 0);
        __builtin_prefetch(xp + xb2 + 256 + ucol, 0, 0);
      }
      const float* rbase = rec + grow * RPAD;
      float4 rz0 = *(const float4*)(rbase + ucol);
      float4 rz1 = *(const float4*)(rbase + ucol + 4);
      float4 rr0 = *(const float4*)(rbase + 128 + ucol);
      float4 rr1 = *(const float4*)(rbase + 128 + ucol + 4);
      float4 rh0 = *(const float4*)(rbase + 256 + ucol);
      float4 rh1 = *(const float4*)(rbase + 256 + ucol + 4);
      float rzv[8] = {rz0.x, rz0.y, rz0.z, rz0.w, rz1.x, rz1.y, rz1.z, rz1.w};
      float rrv[8] = {rr0.x, rr0.y, rr0.z, rr0.w, rr1.x, rr1.y, rr1.z, rr1.w};
      float rhv[8] = {rh0.x, rh0.y, rh0.z, rh0.w, rh1.x, rh1.y, rh1.z, rh1.w};
      v8bf hout;
#pragma unroll
      for (int e = 0; e < 8; e++) {
        const float z  = sigmoidf((float)xz8[e] + rzv[e] + bhz[e]);
        const float r  = sigmoidf((float)xr8[e] + rrv[e] + bhr[e]);
        const float hh = tanhf((float)xh8[e] + r * (rhv[e] + bhh[e]));
        const float hn = z * hreg[e] + (1.0f - z) * hh;
        hreg[e] = hn;
        hout[e] = (__bf16)hn;
      }
      *(v8bf*)(h_bf + grow * UPAD + ucol) = hout;                    // WMMA A for t+1
      *(v8bf*)(out_seq + ((size_t)b * TSEQ + t) * DIN + dir * HU + ucol) = hout;
    }
    __syncthreads();   // h ready for next step
  }
}

// --------------------------- final dense + sigmoid -------------------------
__global__ void dense_out(const __bf16* __restrict__ seq2,
                          const float* __restrict__ Wd,
                          const float* __restrict__ bd,
                          float* __restrict__ out) {
  const int b = blockIdx.x * blockDim.x + threadIdx.x;
  if (b >= BATCH) return;
  float s = bd[0];
  const __bf16* f = seq2 + ((size_t)b * TSEQ + (TSEQ - 1)) * DIN;  // fwd final
  const __bf16* r = seq2 + ((size_t)b * TSEQ + 0) * DIN + HU;      // rev final
  for (int u = 0; u < HU; u++) s += (float)f[u] * Wd[u];
  for (int u = 0; u < HU; u++) s += (float)r[u] * Wd[HU + u];
  out[b] = sigmoidf(s);
}

// ---------------------------------------------------------------------------
extern "C" void kernel_launch(void* const* d_in, const int* in_sizes, int n_in,
                              void* d_out, int out_size, void* d_ws, size_t ws_size,
                              hipStream_t stream) {
  (void)in_sizes; (void)n_in; (void)out_size; (void)ws_size;
  const float* x  = (const float*)d_in[0];  // [256,512,256]
  const float* Ws = (const float*)d_in[1];  // [3,2,256,384]
  const float* Us = (const float*)d_in[2];  // [3,2,128,384]
  const float* bs = (const float*)d_in[3];  // [3,2,2,384]
  const float* Wd = (const float*)d_in[4];  // [256,1]
  const float* bd = (const float*)d_in[5];  // [1]
  float* out = (float*)d_out;               // [256,1] f32

  // bump allocator over workspace (~338 MB total)
  char* ws = (char*)d_ws;
  size_t off = 0;
  auto alloc = [&](size_t bytes) -> void* {
    void* p = ws + off;
    off += (bytes + 255) & ~(size_t)255;
    return p;
  };
  __bf16* seqA = (__bf16*)alloc((size_t)BATCH * TSEQ * DIN * 2);
  __bf16* seqB = (__bf16*)alloc((size_t)BATCH * TSEQ * DIN * 2);
  __bf16* xpF  = (__bf16*)alloc((size_t)BATCH * TSEQ * NOUT * 2);
  __bf16* xpB  = (__bf16*)alloc((size_t)BATCH * TSEQ * NOUT * 2);
  __bf16* Wt   = (__bf16*)alloc((size_t)6 * DIN * NOUT * 2);
  __bf16* Ut   = (__bf16*)alloc((size_t)6 * HU * NOUT * 2);

  conv_weights<<<512, 256, 0, stream>>>(Ws, Us, Wt, Ut);
  conv_f32_bf16<<<4096, 256, 0, stream>>>(x, seqA, (size_t)BATCH * TSEQ * DIN);

  const size_t scan_lds = (size_t)(NOUT + 16) * UPAD * sizeof(__bf16)
                        + (size_t)16 * RPAD * sizeof(float);   // 133,632 B

  for (int l = 0; l < 3; l++) {
    const __bf16* in_seq = (l & 1) ? seqB : seqA;
    __bf16* out_seq      = (l & 1) ? seqA : seqB;
    dim3 gg(6144, 2);                       // 49152 waves, dir in y
    xp_gemm<<<gg, 256, 0, stream>>>(in_seq, Wt, bs, l, xpF, xpB);
    dim3 gs(BATCH / 16, 2);                 // 16 batch-tiles x 2 dirs
    gru_scan<<<gs, 256, scan_lds, stream>>>(xpF, xpB, Ut, bs, l, out_seq);
  }
  dense_out<<<1, 256, 0, stream>>>(seqB, Wd, bd, out);  // layer 2 out = seqB
}
